// FeatureEmbedding_15899968929912
// MI455X (gfx1250) — matrix operand order repaired and checked by reading, not compiled, not run on hardware
//
#include <hip/hip_runtime.h>

typedef __attribute__((ext_vector_type(16))) _Float16 v16h;
typedef __attribute__((ext_vector_type(8)))  _Float16 v8h;
typedef __attribute__((ext_vector_type(8)))  float    v8f;
typedef __attribute__((ext_vector_type(4)))  float    v4f;

#define NROW    262144     // B*F = 256*1024
#define DM      128
#define FF      64
#define WSTRIDE 136        // padded LDS stride in halfs (128 + 8) to spread banks

union V16 { v16h v; v8h h[2]; };

static __device__ inline v16h ldsB16(const _Float16* p) {
    V16 b;
    b.h[0] = *(const v8h*)(p);
    b.h[1] = *(const v8h*)(p + 16);
    return b.v;
}

__global__ __launch_bounds__(256, 1)
void feat_embed_fused(const int*   __restrict__ tokens,
                      const float* __restrict__ rclr,
                      const float* __restrict__ emb,
                      const float* __restrict__ W1,
                      const float* __restrict__ b1,
                      const float* __restrict__ Wl,
                      const float* __restrict__ bl,
                      const float* __restrict__ Wp,
                      const float* __restrict__ bp,
                      const float* __restrict__ gamma,
                      const float* __restrict__ beta,
                      float*       __restrict__ out)
{
    extern __shared__ _Float16 smem[];
    _Float16* W1t   = smem;            // [128][WSTRIDE]  transposed: W1t[n][k]
    _Float16* Wlt   = smem + 17408;    // [64][WSTRIDE]
    _Float16* Wpt   = smem + 26112;    // [64][WSTRIDE]
    _Float16* xnorm = smem + 34816;    // [8 waves][16][WSTRIDE]

    const int tid  = threadIdx.x;
    const int lane = tid & 31;
    const int wave = tid >> 5;
    const int blk  = blockIdx.x;
    const int row0 = blk * 128 + wave * 16;

    // ---- outputs 0/1: feature_mask and randomized_tokens (augmentation ~ identity)
    if (tid < 128) {
        int rg = blk * 128 + tid;
        int t  = tokens[rg];
        out[rg]        = (t != 0) ? 1.0f : 0.0f;
        out[NROW + rg] = (float)t;
    }

    // ---- cooperative preload of weights into LDS (f32 -> f16, transposed)
    for (int i = tid; i < 128 * 128; i += 256) {
        int k = i >> 7, n = i & 127;
        W1t[n * WSTRIDE + k] = (_Float16)W1[i];
    }
    for (int i = tid; i < 128 * 64; i += 256) {
        int k = i >> 6, n = i & 63;
        Wlt[n * WSTRIDE + k] = (_Float16)Wl[i];
        Wpt[n * WSTRIDE + k] = (_Float16)Wp[i];
    }
    __syncthreads();

    const int m   = lane & 15;   // row (A) / col (B,C) index within 16-tile
    const int grp = lane >> 4;   // lane half-group

    // ---- A fragments: gather this wave's 16 embedding rows, f32 -> f16
    // 16-bit A 16x32 layout: lanes 0-15 hold K {0..7,16..23}, lanes 16-31 hold K {8..15,24..31}
    int tok = tokens[row0 + m];
    const float* erow = emb + (long)tok * 128;
    V16 a[4];
    #pragma unroll
    for (int kc = 0; kc < 4; ++kc) {
        int kb = kc * 32 + grp * 8;
        v4f f0 = *(const v4f*)(erow + kb);
        v4f f1 = *(const v4f*)(erow + kb + 4);
        v4f f2 = *(const v4f*)(erow + kb + 16);
        v4f f3 = *(const v4f*)(erow + kb + 20);
        #pragma unroll
        for (int j = 0; j < 4; ++j) {
            a[kc].h[0][j]     = (_Float16)f0[j];
            a[kc].h[0][4 + j] = (_Float16)f1[j];
            a[kc].h[1][j]     = (_Float16)f2[j];
            a[kc].h[1][4 + j] = (_Float16)f3[j];
        }
    }

    // ---- GEMM1: [16x128] x [128x128] via 8 N-tiles x 4 K-steps of wmma f16
    v8f acc[8];
    #pragma unroll
    for (int nt = 0; nt < 8; ++nt) {
        v8f c = {0.f, 0.f, 0.f, 0.f, 0.f, 0.f, 0.f, 0.f};
        #pragma unroll
        for (int kc = 0; kc < 4; ++kc) {
            // B 32x16 layout: lane n = lane&15, lanes 0-15 K {0..7,16..23}, lanes 16-31 K {8..15,24..31}
            v16h bfrag = ldsB16(W1t + (nt * 16 + m) * WSTRIDE + kc * 32 + grp * 8);
            c = __builtin_amdgcn_wmma_f32_16x16x32_f16(
                    false, a[kc].v, false, bfrag, (short)0, c, false, false);
        }
        acc[nt] = c;
    }

    // ---- bias + relu + per-row scale (rclr; shuffled_mask == 0 in our variant)
    float sc[8];
    #pragma unroll
    for (int r = 0; r < 8; ++r) sc[r] = rclr[row0 + grp * 8 + r];

    #pragma unroll
    for (int nt = 0; nt < 8; ++nt) {
        float bias = b1[nt * 16 + m];
        #pragma unroll
        for (int r = 0; r < 8; ++r) {
            float x = acc[nt][r] + bias;
            acc[nt][r] = fmaxf(x, 0.0f) * sc[r];
        }
    }

    // ---- LayerNorm over 128 cols; row values live across 16 lanes x 8 nt-tiles
    float mu[8], rs[8];
    #pragma unroll
    for (int r = 0; r < 8; ++r) {
        float s = 0.f, q = 0.f;
        #pragma unroll
        for (int nt = 0; nt < 8; ++nt) { float v = acc[nt][r]; s += v; q += v * v; }
        #pragma unroll
        for (int o = 1; o < 16; o <<= 1) {
            s += __shfl_xor(s, o, 16);
            q += __shfl_xor(q, o, 16);
        }
        float mean = s * (1.0f / 128.0f);
        float var  = q * (1.0f / 128.0f) - mean * mean;
        mu[r] = mean;
        rs[r] = rsqrtf(var + 1e-3f);
    }

    // ---- normalize + gamma/beta, stash f16 rows in LDS for the C->A transpose
    _Float16* xw = xnorm + wave * (16 * WSTRIDE);
    #pragma unroll
    for (int nt = 0; nt < 8; ++nt) {
        float g  = gamma[nt * 16 + m];
        float be = beta[nt * 16 + m];
        #pragma unroll
        for (int r = 0; r < 8; ++r) {
            float y = (acc[nt][r] - mu[r]) * rs[r] * g + be;
            xw[(grp * 8 + r) * WSTRIDE + nt * 16 + m] = (_Float16)y;
        }
    }
    // per-wave region; DS ops from the same wave are in-order, no barrier needed

    // ---- reload normalized activations as A fragments
    const _Float16* xr = xw + m * WSTRIDE;
    #pragma unroll
    for (int kc = 0; kc < 4; ++kc) {
        int kb = kc * 32 + grp * 8;
        a[kc].h[0] = *(const v8h*)(xr + kb);
        a[kc].h[1] = *(const v8h*)(xr + kb + 16);
    }

    // ---- GEMM2: two [16x128] x [128x64] products (loadings + pca), fused stores
    float* out_e = out + 2L * NROW;                   // output_embeddings [NROW][64]
    float* out_p = out + 2L * NROW + (long)NROW * FF; // output_regression [NROW][64]

    #pragma unroll
    for (int nt2 = 0; nt2 < 4; ++nt2) {
        v8f cl = {0.f, 0.f, 0.f, 0.f, 0.f, 0.f, 0.f, 0.f};
        v8f cp = {0.f, 0.f, 0.f, 0.f, 0.f, 0.f, 0.f, 0.f};
        #pragma unroll
        for (int kc = 0; kc < 4; ++kc) {
            v16h bfl = ldsB16(Wlt + (nt2 * 16 + m) * WSTRIDE + kc * 32 + grp * 8);
            cl = __builtin_amdgcn_wmma_f32_16x16x32_f16(
                     false, a[kc].v, false, bfl, (short)0, cl, false, false);
            v16h bfp = ldsB16(Wpt + (nt2 * 16 + m) * WSTRIDE + kc * 32 + grp * 8);
            cp = __builtin_amdgcn_wmma_f32_16x16x32_f16(
                     false, a[kc].v, false, bfp, (short)0, cp, false, false);
        }
        float bll = bl[nt2 * 16 + m];
        float bpp = bp[nt2 * 16 + m];
        #pragma unroll
        for (int r = 0; r < 8; ++r) {
            long rg = (long)row0 + grp * 8 + r;
            out_e[rg * FF + nt2 * 16 + m] = cl[r] + bll;
            out_p[rg * FF + nt2 * 16 + m] = cp[r] + bpp;
        }
    }
}

extern "C" void kernel_launch(void* const* d_in, const int* in_sizes, int n_in,
                              void* d_out, int out_size, void* d_ws, size_t ws_size,
                              hipStream_t stream) {
    (void)in_sizes; (void)n_in; (void)out_size; (void)d_ws; (void)ws_size;
    const int*   tokens = (const int*)  d_in[0];
    const float* rclr   = (const float*)d_in[1];
    const float* emb    = (const float*)d_in[2];
    const float* W1     = (const float*)d_in[3];
    const float* b1     = (const float*)d_in[4];
    const float* Wl     = (const float*)d_in[5];
    const float* bl     = (const float*)d_in[6];
    const float* Wp     = (const float*)d_in[7];
    const float* bp     = (const float*)d_in[8];
    const float* gamma  = (const float*)d_in[9];
    const float* beta   = (const float*)d_in[10];
    float* out = (float*)d_out;

    // LDS: W1t 34816B + Wlt 17408B + Wpt 17408B + xnorm 34816B = 104448 bytes
    size_t smem_bytes = (size_t)(17408 + 8704 + 8704 + 17408) * sizeof(_Float16);
    dim3 grid(NROW / 128);
    dim3 block(256);
    feat_embed_fused<<<grid, block, smem_bytes, stream>>>(
        tokens, rclr, emb, W1, b1, Wl, bl, Wp, bp, gamma, beta, out);
}
